// Gate_28192165331299
// MI455X (gfx1250) — matrix operand order repaired and checked by reading, not compiled, not run on hardware
//
#include <hip/hip_runtime.h>

typedef __attribute__((ext_vector_type(16))) __bf16 v16bf;
typedef __attribute__((ext_vector_type(2)))  __bf16 v2bf;
typedef __attribute__((ext_vector_type(8)))  float  v8f;

#define D_DIM 256
#define E_DIM 64
#define WROW 264      // bf16 elements per expert row in LDS (256 + 8 pad)
#define LROW 65       // float elements per token row for logits (64 + 1 pad)
#define TOK_PER_BLK 128

__global__ __launch_bounds__(256)
void gate_wmma_kernel(const float* __restrict__ x, const float* __restrict__ W,
                      float* __restrict__ outw, int* __restrict__ outi, int T)
{
    // 33792 bytes: holds bf16 W during GEMM, reused for fp32 logits afterwards
    __shared__ __align__(16) unsigned char smem[E_DIM * WROW * 2];
    __bf16* wlds = reinterpret_cast<__bf16*>(smem);
    float*  llds = reinterpret_cast<float*>(smem);

    const int tid  = threadIdx.x;
    const int wave = tid >> 5;
    const int lane = tid & 31;
    const int r    = lane & 15;   // token row (A) / expert col (B/C/D) within tile
    const int h    = lane >> 4;   // lane half

    // ---- Stage W (fp32 -> bf16) into LDS, padded rows ----
    const float2* W2 = reinterpret_cast<const float2*>(W);
    for (int i = tid; i < E_DIM * (D_DIM / 2); i += 256) {
        int row = i >> 7;          // / (D_DIM/2)
        int p   = i & 127;
        float2 w2 = W2[row * (D_DIM / 2) + p];
        v2bf pk;
        pk[0] = (__bf16)w2.x;
        pk[1] = (__bf16)w2.y;
        *reinterpret_cast<v2bf*>(wlds + row * WROW + 2 * p) = pk;
    }
    __syncthreads();

    // ---- GEMM: each wave = 16 tokens x 64 experts, K=256 in 8 chunks of 32 ----
    const int tok0 = blockIdx.x * TOK_PER_BLK + wave * 16 + r;
    const int tokc = tok0 < T ? tok0 : (T - 1);          // clamp (no divergence)
    const float* xr = x + (size_t)tokc * D_DIM;

    v8f c0 = {}, c1 = {}, c2 = {}, c3 = {};

    for (int kc = 0; kc < 8; ++kc) {
        const int kb = kc * 32;
        // A fragment: contiguous k order -> a[0..15]=K(16h..16h+15)
        float4 f0 = *reinterpret_cast<const float4*>(xr + kb + 16 * h);
        float4 f1 = *reinterpret_cast<const float4*>(xr + kb + 16 * h + 4);
        float4 f2 = *reinterpret_cast<const float4*>(xr + kb + 16 * h + 8);
        float4 f3 = *reinterpret_cast<const float4*>(xr + kb + 16 * h + 12);
        v16bf a;
        a[0]  = (__bf16)f0.x; a[1]  = (__bf16)f0.y; a[2]  = (__bf16)f0.z; a[3]  = (__bf16)f0.w;
        a[4]  = (__bf16)f1.x; a[5]  = (__bf16)f1.y; a[6]  = (__bf16)f1.z; a[7]  = (__bf16)f1.w;
        a[8]  = (__bf16)f2.x; a[9]  = (__bf16)f2.y; a[10] = (__bf16)f2.z; a[11] = (__bf16)f2.w;
        a[12] = (__bf16)f3.x; a[13] = (__bf16)f3.y; a[14] = (__bf16)f3.z; a[15] = (__bf16)f3.w;

        // B fragments: lane (n=r, half h) reads 16 contiguous bf16 = W[expert][kb+16h ..]
        union BU { uint4 u[2]; v16bf v; } bu;
#define LOADB(nt, dst) {                                                            \
        const uint4* bp = reinterpret_cast<const uint4*>(                           \
            wlds + ((nt) * 16 + r) * WROW + kb + 16 * h);                           \
        bu.u[0] = bp[0]; bu.u[1] = bp[1]; dst = bu.v; }

        v16bf b;
        LOADB(0, b);
        c0 = __builtin_amdgcn_wmma_f32_16x16x32_bf16(false, a, false, b, (short)0, c0, false, false);
        LOADB(1, b);
        c1 = __builtin_amdgcn_wmma_f32_16x16x32_bf16(false, a, false, b, (short)0, c1, false, false);
        LOADB(2, b);
        c2 = __builtin_amdgcn_wmma_f32_16x16x32_bf16(false, a, false, b, (short)0, c2, false, false);
        LOADB(3, b);
        c3 = __builtin_amdgcn_wmma_f32_16x16x32_bf16(false, a, false, b, (short)0, c3, false, false);
#undef LOADB
    }

    // ---- Spill logits to LDS (reuse W buffer after all waves finish GEMM) ----
    __syncthreads();
#pragma unroll
    for (int v = 0; v < 8; ++v) {
        // D layout: VGPR v, lane (n=r, half h) holds element [M = v + 8h][N = n]
        int row = wave * 16 + v + 8 * h;
        llds[row * LROW + r +  0] = c0[v];
        llds[row * LROW + r + 16] = c1[v];
        llds[row * LROW + r + 32] = c2[v];
        llds[row * LROW + r + 48] = c3[v];
    }
    __syncthreads();

    // ---- Softmax + grouped top-k: one thread per token ----
    if (tid < TOK_PER_BLK) {
        int tg = blockIdx.x * TOK_PER_BLK + tid;
        if (tg < T) {
            const float* l = llds + tid * LROW;
            float g0 = -__builtin_inff(), g1 = -__builtin_inff();
            for (int i = 0;  i < 32; ++i) g0 = fmaxf(g0, l[i]);
            for (int i = 32; i < 64; ++i) g1 = fmaxf(g1, l[i]);
            float m = fmaxf(g0, g1);
            float s = 0.0f;
            for (int i = 0; i < 64; ++i) s += __expf(l[i] - m);
            float inv = 1.0f / s;                 // ROUTE_SCALE == 1.0
            int base = (g0 >= g1) ? 0 : 32;       // top-1 group, ties -> group 0
            unsigned mask = 0u;
#pragma unroll
            for (int k = 0; k < 4; ++k) {
                float best = -__builtin_inff(); int bi = 0;
                for (int i = 0; i < 32; ++i) {
                    if (!((mask >> i) & 1u)) {
                        float v = l[base + i];
                        if (v > best) { best = v; bi = i; }  // ties -> lowest index
                    }
                }
                mask |= (1u << bi);
                outw[(size_t)tg * 4 + k] = __expf(best - m) * inv;
                outi[(size_t)tg * 4 + k] = base + bi;
            }
        }
    }
}

extern "C" void kernel_launch(void* const* d_in, const int* in_sizes, int n_in,
                              void* d_out, int out_size, void* d_ws, size_t ws_size,
                              hipStream_t stream) {
    const float* x = (const float*)d_in[0];   // [T, 256] f32
    const float* W = (const float*)d_in[1];   // [64, 256] f32
    int T = in_sizes[0] / D_DIM;
    float* outw = (float*)d_out;                       // [T,4] f32 weights
    int*   outi = (int*)d_out + (size_t)T * 4;         // [T,4] i32 indices (raw bits)
    int blocks = (T + TOK_PER_BLK - 1) / TOK_PER_BLK;
    gate_wmma_kernel<<<blocks, 256, 0, stream>>>(x, W, outw, outi, T);
}
